// ClusterForecasting_17944373362977
// MI455X (gfx1250) — matrix operand order repaired and checked by reading, not compiled, not run on hardware
//
#include <hip/hip_runtime.h>
#include <hip/hip_bf16.h>

typedef __attribute__((ext_vector_type(16))) _Float16 v16h;
typedef __attribute__((ext_vector_type(8)))  _Float16 v8h;
typedef __attribute__((ext_vector_type(8)))  float    v8f;

namespace {
constexpr int kB = 64, kS = 256, kIN = 32, kD = 512, kH = 8, kL = 2, kDFF = 2048, kC = 16, kDH = 64;
constexpr int kM = kB * kS;  // 16384 activation rows
}

// ---------------------------------------------------------------------------
// f32 -> f16 elementwise convert
// ---------------------------------------------------------------------------
__global__ __launch_bounds__(256) void cvt_kernel(const float* __restrict__ in,
                                                  _Float16* __restrict__ out, long n) {
  long i = (long)blockIdx.x * blockDim.x + threadIdx.x;
  if (i < n) out[i] = (_Float16)in[i];
}

// f32 [mats,K,N] -> f16 [mats,N,K]  (weights become K-contiguous for WMMA B-side)
__global__ __launch_bounds__(256) void cvt_t_kernel(const float* __restrict__ in,
                                                    _Float16* __restrict__ out,
                                                    int K, int N, long total) {
  long j = (long)blockIdx.x * blockDim.x + threadIdx.x;
  if (j >= total) return;
  long kn = (long)K * N;
  long mat = j / kn;
  long rem = j - mat * kn;
  long n = rem / K;
  long k = rem - n * K;
  out[j] = (_Float16)in[mat * kn + k * N + n];
}

// v [B,S,H*DH] f16 -> vT [B,H,DH,S] f16 (K-contiguous B operand for P@V)
__global__ __launch_bounds__(256) void vperm_kernel(const _Float16* __restrict__ v,
                                                    _Float16* __restrict__ vt) {
  long i = (long)blockIdx.x * blockDim.x + threadIdx.x;  // over B*H*DH*S
  if (i >= (long)kM * kD) return;
  int s = (int)(i & (kS - 1));
  long j = i >> 8;
  int dh = (int)(j & (kDH - 1));
  long j2 = j >> 6;
  int h = (int)(j2 & (kH - 1));
  long b = j2 >> 3;
  vt[i] = v[((b * kS + s) * kD) + h * kDH + dh];
}

// ---------------------------------------------------------------------------
// WMMA fragment loads (K-contiguous, 16B-aligned by construction)
// A 16x32 f16: lane%16 = row, K = 16*(lane>=16) + elem  (16 contiguous halves)
// B 32x16 f16: lane%16 = col, K = 16*(lane>=16) + elem  (16 contiguous halves)
// ---------------------------------------------------------------------------
__device__ inline v16h ldA(const _Float16* p) {  // p already offset by (lane>=16)*16
  v8h lo = *(const v8h*)p;
  v8h hi = *(const v8h*)(p + 8);
  return __builtin_shufflevector(lo, hi, 0, 1, 2, 3, 4, 5, 6, 7, 8, 9, 10, 11, 12, 13, 14, 15);
}
__device__ inline v16h ldB(const _Float16* p) {  // p already offset by (lane>=16)*16
  v8h lo = *(const v8h*)p;
  v8h hi = *(const v8h*)(p + 8);
  return __builtin_shufflevector(lo, hi, 0, 1, 2, 3, 4, 5, 6, 7, 8, 9, 10, 11, 12, 13, 14, 15);
}

// ---------------------------------------------------------------------------
// Batched WMMA GEMM, both operands K-major:
//   A(m,k) = A[m*aSM + k]        B(k,n) = B[n*bSN + k]
//   C[batch](m,n) = act( A@B + bias[n] + res(m,n) )
// Each wave computes a 16x32 output slab (one A fragment, two B tiles).
// Batch offset = (batch/batchMod)*X1 + (batch%batchMod)*X2.
// ---------------------------------------------------------------------------
__global__ __launch_bounds__(256) void wmma_gemm(
    const _Float16* __restrict__ A, long aSM, long aB1, long aB2,
    const _Float16* __restrict__ Bm, long bSN, long bB1, long bB2,
    const float* __restrict__ bias,
    const float* __restrict__ res, long rLD, long rB1, long rB2,
    float* __restrict__ Cf, _Float16* __restrict__ Ch, long cLD, long cB1, long cB2,
    int tilesM, int K, int slabsTotal, int batchMod, int relu) {
  int wave = threadIdx.x >> 5;
  int lane = threadIdx.x & 31;
  int gw = blockIdx.x * 8 + wave;            // slab id (wave-uniform)
  if (gw >= slabsTotal) return;              // uniform exit: EXEC all-ones for WMMA
  int batch = blockIdx.y;
  long ob1 = batch / batchMod;
  long ob2 = batch % batchMod;

  int tm = gw % tilesM;
  int tn2 = gw / tilesM;
  int row0 = tm << 4;
  int col0 = tn2 << 5;                       // 32-wide slab
  int r = lane & 15;
  int hi = lane >> 4;

  const _Float16* ap = A + ob1 * aB1 + ob2 * aB2 + (long)(row0 + r) * aSM + (hi << 4);
  const _Float16* bp0 = Bm + ob1 * bB1 + ob2 * bB2 + (long)(col0 + r) * bSN + (hi << 4);
  const _Float16* bp1 = bp0 + 16 * bSN;

  v8f acc0 = {};
  v8f acc1 = {};
#pragma unroll 2
  for (int k0 = 0; k0 < K; k0 += 32) {
    __builtin_prefetch(ap + k0 + 64, 0, 0);    // global_prefetch_b8: next K tiles
    __builtin_prefetch(bp0 + k0 + 64, 0, 0);
    v16h af = ldA(ap + k0);
    v16h bf0 = ldB(bp0 + k0);
    v16h bf1 = ldB(bp1 + k0);
    acc0 = __builtin_amdgcn_wmma_f32_16x16x32_f16(false, af, false, bf0, (short)0, acc0, false, false);
    acc1 = __builtin_amdgcn_wmma_f32_16x16x32_f16(false, af, false, bf1, (short)0, acc1, false, false);
  }

  long cOff = ob1 * cB1 + ob2 * cB2;
  int colg0 = col0 + r;
  int colg1 = colg0 + 16;
  float bv0 = bias ? bias[colg0] : 0.0f;
  float bv1 = bias ? bias[colg1] : 0.0f;
  const float* rp = res ? res + ob1 * rB1 + ob2 * rB2 : nullptr;
#pragma unroll
  for (int i = 0; i < 8; ++i) {
    int rowg = row0 + (hi << 3) + i;
    float v0 = acc0[i] + bv0;
    float v1 = acc1[i] + bv1;
    if (rp) {
      v0 += rp[(long)rowg * rLD + colg0];
      v1 += rp[(long)rowg * rLD + colg1];
    }
    if (relu) {
      v0 = fmaxf(v0, 0.0f);
      v1 = fmaxf(v1, 0.0f);
    }
    long c0 = cOff + (long)rowg * cLD + colg0;
    long c1 = cOff + (long)rowg * cLD + colg1;
    if (Cf) { Cf[c0] = v0; Cf[c1] = v1; }
    if (Ch) { Ch[c0] = (_Float16)v0; Ch[c1] = (_Float16)v1; }
  }
}

// ---------------------------------------------------------------------------
// LayerNorm over rows of 512; writes f32 (residual stream) and f16 (GEMM input)
// ---------------------------------------------------------------------------
__global__ __launch_bounds__(256) void ln_kernel(const float* __restrict__ x,
                                                 const float* __restrict__ sc,
                                                 const float* __restrict__ bi,
                                                 float* __restrict__ yf,
                                                 _Float16* __restrict__ yh) {
  __shared__ float red[256];
  long row = blockIdx.x;
  const float* xr = x + row * kD;
  int t = threadIdx.x;
  float a = xr[t], b = xr[t + 256];
  red[t] = a + b;
  __syncthreads();
  for (int s2 = 128; s2 > 0; s2 >>= 1) {
    if (t < s2) red[t] += red[t + s2];
    __syncthreads();
  }
  float mean = red[0] * (1.0f / kD);
  __syncthreads();
  float da = a - mean, db = b - mean;
  red[t] = da * da + db * db;
  __syncthreads();
  for (int s2 = 128; s2 > 0; s2 >>= 1) {
    if (t < s2) red[t] += red[t + s2];
    __syncthreads();
  }
  float rstd = rsqrtf(red[0] * (1.0f / kD) + 1e-5f);
  float y0 = da * rstd * sc[t] + bi[t];
  float y1 = db * rstd * sc[t + 256] + bi[t + 256];
  yf[row * kD + t] = y0;
  yf[row * kD + t + 256] = y1;
  yh[row * kD + t] = (_Float16)y0;
  yh[row * kD + t + 256] = (_Float16)y1;
}

// ---------------------------------------------------------------------------
// In-place softmax over f16 rows of length 256 (attention scores * 1/sqrt(DH))
// ---------------------------------------------------------------------------
__global__ __launch_bounds__(256) void softmax_kernel(_Float16* __restrict__ p, float scale) {
  __shared__ float red[256];
  long row = blockIdx.x;
  _Float16* pr = p + row * kS;
  int t = threadIdx.x;
  float v = (float)pr[t] * scale;
  red[t] = v;
  __syncthreads();
  for (int s2 = 128; s2 > 0; s2 >>= 1) {
    if (t < s2) red[t] = fmaxf(red[t], red[t + s2]);
    __syncthreads();
  }
  float mx = red[0];
  __syncthreads();
  float e = __expf(v - mx);
  red[t] = e;
  __syncthreads();
  for (int s2 = 128; s2 > 0; s2 >>= 1) {
    if (t < s2) red[t] += red[t + s2];
    __syncthreads();
  }
  pr[t] = (_Float16)(e / red[0]);
}

// ---------------------------------------------------------------------------
// Clustering epilogue kernels
// ---------------------------------------------------------------------------
__global__ __launch_bounds__(256) void init_kernel(float* __restrict__ out, float* __restrict__ cm) {
  int t = threadIdx.x;
  if (t == 0) out[0] = 0.0f;
  cm[t] = 0.0f;
}

// One thread per (s,b): dist row from gram, select 16 NN, counts, tot_sum, assigned label.
__global__ __launch_bounds__(256) void knn_kernel(const float* __restrict__ gram,
                                                  const int* __restrict__ y,
                                                  int* __restrict__ assigned_i,
                                                  float* __restrict__ out) {
  int idx = blockIdx.x * blockDim.x + threadIdx.x;  // idx = s*B + b (matches reference flatten)
  if (idx >= kS * kB) return;
  int s = idx >> 6;
  int b = idx & 63;
  const float* g = gram + (long)s * kB * kB;
  float gbb = g[b * kB + b];
  float dist[kB];
  float dsum = 0.0f;
  for (int c = 0; c < kB; ++c) {
    float d = gbb + g[c * kB + c] - 2.0f * g[b * kB + c];
    dist[c] = d;
    dsum += d;
  }
  // 16 nearest (softmax of -dist is monotone, so top_k(p) == 16 smallest dists; stable ties)
  unsigned long long used = 0ull;
  int knn[kC];
  for (int j = 0; j < kC; ++j) {
    float best = 3.4e38f;
    int bi = 0;
    for (int c = 0; c < kB; ++c) {
      if (!((used >> c) & 1ull) && dist[c] < best) { best = dist[c]; bi = c; }
    }
    used |= (1ull << bi);
    knn[j] = bi;
  }
  int counts = 0;
  for (int j = 0; j < kC; ++j) counts += (knn[j] == j);
  atomicAdd(&out[0], dsum * (float)counts);
  int cnt[kC];
  for (int c = 0; c < kC; ++c) cnt[c] = 0;
  for (int j = 0; j < kC; ++j) {
    int lab = y[(long)knn[j] * kS + s] & (kC - 1);
    cnt[lab]++;
  }
  int bestc = 0, bc = cnt[0];
  for (int c = 1; c < kC; ++c)
    if (cnt[c] > bc) { bc = cnt[c]; bestc = c; }
  assigned_i[idx] = bestc;
  out[2 + idx] = (float)bestc;
}

__global__ __launch_bounds__(256) void cm_kernel(const int* __restrict__ assigned_i,
                                                 const int* __restrict__ y,
                                                 float* __restrict__ cm) {
  int i = blockIdx.x * blockDim.x + threadIdx.x;
  if (i >= kM) return;
  int a = assigned_i[i] & (kC - 1);
  int t = y[i] & (kC - 1);
  atomicAdd(&cm[a * kC + t], 1.0f);
}

__global__ void ari_kernel(const float* __restrict__ cm, float* __restrict__ out) {
  if (threadIdx.x != 0 || blockIdx.x != 0) return;
  float sij = 0.0f, rs[kC], cs[kC];
  for (int i = 0; i < kC; ++i) { rs[i] = 0.0f; cs[i] = 0.0f; }
  for (int a = 0; a < kC; ++a)
    for (int t = 0; t < kC; ++t) {
      float m = cm[a * kC + t];
      sij += m * (m - 1.0f) * 0.5f;
      rs[a] += m;
      cs[t] += m;
    }
  float sa = 0.0f, sb = 0.0f;
  for (int i = 0; i < kC; ++i) {
    sa += rs[i] * (rs[i] - 1.0f) * 0.5f;
    sb += cs[i] * (cs[i] - 1.0f) * 0.5f;
  }
  float n = (float)kM;
  float ex = sa * sb / (n * (n - 1.0f) * 0.5f);
  float mx = 0.5f * (sa + sb);
  out[1] = (sij - ex) / (mx - ex);
}

// ---------------------------------------------------------------------------
// Host-side helpers
// ---------------------------------------------------------------------------
static void gemm(hipStream_t st,
                 const _Float16* A, long aSM, long aB1, long aB2,
                 const _Float16* B, long bSN, long bB1, long bB2,
                 const float* bias,
                 const float* res, long rLD, long rB1, long rB2,
                 float* Cf, _Float16* Ch, long cLD, long cB1, long cB2,
                 int M, int N, int K, int batches, int batchMod, int relu) {
  int tilesM = M / 16;
  int slabs = tilesM * (N / 32);  // all N in this net are multiples of 32
  dim3 grid((slabs + 7) / 8, batches, 1);
  wmma_gemm<<<grid, 256, 0, st>>>(A, aSM, aB1, aB2, B, bSN, bB1, bB2, bias,
                                  res, rLD, rB1, rB2, Cf, Ch, cLD, cB1, cB2,
                                  tilesM, K, slabs, batchMod, relu);
}

static void cvt(hipStream_t st, const float* src, _Float16* dst, long n) {
  cvt_kernel<<<(unsigned)((n + 255) / 256), 256, 0, st>>>(src, dst, n);
}

static void cvtT(hipStream_t st, const float* src, _Float16* dst, int K, int N, int mats) {
  long total = (long)mats * K * N;
  cvt_t_kernel<<<(unsigned)((total + 255) / 256), 256, 0, st>>>(src, dst, K, N, total);
}

extern "C" void kernel_launch(void* const* d_in, const int* in_sizes, int n_in,
                              void* d_out, int out_size, void* d_ws, size_t ws_size,
                              hipStream_t stream) {
  const float* x = (const float*)d_in[0];
  const float* W_emb = (const float*)d_in[1];
  const float* b_emb = (const float*)d_in[2];
  const float* Wq = (const float*)d_in[3];
  const float* bq = (const float*)d_in[4];
  const float* Wk = (const float*)d_in[5];
  const float* bk = (const float*)d_in[6];
  const float* Wv = (const float*)d_in[7];
  const float* bv = (const float*)d_in[8];
  const float* Wo = (const float*)d_in[9];
  const float* bo = (const float*)d_in[10];
  const float* ln1_s = (const float*)d_in[11];
  const float* ln1_b = (const float*)d_in[12];
  const float* W1 = (const float*)d_in[13];
  const float* b1 = (const float*)d_in[14];
  const float* W2 = (const float*)d_in[15];
  const float* b2 = (const float*)d_in[16];
  const float* ln2_s = (const float*)d_in[17];
  const float* ln2_b = (const float*)d_in[18];
  const float* e1_w = (const float*)d_in[19];
  const float* e1_b = (const float*)d_in[20];
  const float* e2_w = (const float*)d_in[21];
  const float* e2_b = (const float*)d_in[22];
  const float* d1_w = (const float*)d_in[23];
  const float* d1_b = (const float*)d_in[24];
  const float* d2_w = (const float*)d_in[25];
  const float* d2_b = (const float*)d_in[26];
  const int* y = (const int*)d_in[27];
  float* out = (float*)d_out;

  // ---- workspace carve (256B aligned) ----
  char* wsb = (char*)d_ws;
  size_t cur = 0;
  auto alloc = [&](size_t bytes) -> void* {
    void* p = (void*)(wsb + cur);
    cur = (cur + bytes + 255) & ~(size_t)255;
    return p;
  };
  _Float16* x16 = (_Float16*)alloc((size_t)kM * kIN * 2);
  // all weights stored TRANSPOSED: [N,K] f16, K-contiguous for WMMA B-side
  _Float16* wembT = (_Float16*)alloc((size_t)kIN * kD * 2);
  _Float16* wqT = (_Float16*)alloc((size_t)kL * kD * kD * 2);
  _Float16* wkT = (_Float16*)alloc((size_t)kL * kD * kD * 2);
  _Float16* wvT = (_Float16*)alloc((size_t)kL * kD * kD * 2);
  _Float16* woT = (_Float16*)alloc((size_t)kL * kD * kD * 2);
  _Float16* w1T = (_Float16*)alloc((size_t)kL * kD * kDFF * 2);
  _Float16* w2T = (_Float16*)alloc((size_t)kL * kDFF * kD * 2);
  _Float16* e1T = (_Float16*)alloc((size_t)kD * 128 * 2);
  _Float16* e2T = (_Float16*)alloc((size_t)128 * kD * 2);
  _Float16* d1T = (_Float16*)alloc((size_t)kD * 128 * 2);
  _Float16* d2T = (_Float16*)alloc((size_t)128 * kD * 2);
  float* hf = (float*)alloc((size_t)kM * kD * 4);     // residual stream (f32)
  _Float16* hh = (_Float16*)alloc((size_t)kM * kD * 2);
  float* tmpf = (float*)alloc((size_t)kM * kD * 4);   // pre-LN sum
  _Float16* o16 = (_Float16*)alloc((size_t)kM * kD * 2);
  // region A: q/k/v during attention (vT overwrites q after scores); tail buffers afterwards
  char* regA = (char*)alloc((size_t)3 * kM * kD * 2);
  _Float16* q16 = (_Float16*)regA;
  _Float16* k16 = q16 + (size_t)kM * kD;
  _Float16* v16 = k16 + (size_t)kM * kD;
  _Float16* vT = q16;                              // [B,H,DH,S]; q is dead after scores
  _Float16* g1h = (_Float16*)regA;                 // [M,128]
  _Float16* ench = g1h + (size_t)kM * 128;         // [M,512]
  _Float16* g2h = ench + (size_t)kM * kD;          // [M,128]
  _Float16* z16 = g2h + (size_t)kM * 128;          // [M,512]
  float* gram = (float*)(z16 + (size_t)kM * kD);   // [S,64,64]
  // region B: attention scores/probs (f16) union FFN hidden (f16) -- both 64MiB
  char* regB = (char*)alloc((size_t)kB * kH * kS * kS * 2);
  _Float16* s16 = (_Float16*)regB;
  _Float16* f1h = (_Float16*)regB;
  int* assigned_i = (int*)alloc((size_t)kM * 4);
  float* cmbuf = (float*)alloc(256 * 4);

  // ---- conversions: x plain, weights transposed to [N,K] ----
  cvt(stream, x, x16, (long)kM * kIN);
  cvtT(stream, W_emb, wembT, kIN, kD, 1);
  cvtT(stream, Wq, wqT, kD, kD, kL);
  cvtT(stream, Wk, wkT, kD, kD, kL);
  cvtT(stream, Wv, wvT, kD, kD, kL);
  cvtT(stream, Wo, woT, kD, kD, kL);
  cvtT(stream, W1, w1T, kD, kDFF, kL);
  cvtT(stream, W2, w2T, kDFF, kD, kL);
  cvtT(stream, e1_w, e1T, kD, 128, 1);
  cvtT(stream, e2_w, e2T, 128, kD, 1);
  cvtT(stream, d1_w, d1T, kD, 128, 1);
  cvtT(stream, d2_w, d2T, 128, kD, 1);

  // ---- embedding: h = x @ W_emb + b_emb ----
  gemm(stream, x16, kIN, 0, 0, wembT, kIN, 0, 0, b_emb,
       nullptr, 0, 0, 0, hf, hh, kD, 0, 0, kM, kD, kIN, 1, 1, 0);

  const long SD = (long)kS * kD;        // per-b stride in activations
  const long HSS = (long)kH * kS * kS;  // per-b stride in scores
  const long SS = (long)kS * kS;
  const long HDS = (long)kH * kDH * kS; // per-b stride in vT
  const long DS = (long)kDH * kS;       // per-h stride in vT

  for (int l = 0; l < kL; ++l) {
    const size_t wOff = (size_t)l * kD * kD;
    // q,k,v projections
    gemm(stream, hh, kD, 0, 0, wqT + wOff, kD, 0, 0, bq + l * kD,
         nullptr, 0, 0, 0, nullptr, q16, kD, 0, 0, kM, kD, kD, 1, 1, 0);
    gemm(stream, hh, kD, 0, 0, wkT + wOff, kD, 0, 0, bk + l * kD,
         nullptr, 0, 0, 0, nullptr, k16, kD, 0, 0, kM, kD, kD, 1, 1, 0);
    gemm(stream, hh, kD, 0, 0, wvT + wOff, kD, 0, 0, bv + l * kD,
         nullptr, 0, 0, 0, nullptr, v16, kD, 0, 0, kM, kD, kD, 1, 1, 0);
    // scores[b,h] = Q[b,:,h,:] @ K[b,:,h,:]^T  (k16 is already K-contiguous as B)
    gemm(stream, q16, kD, SD, kDH,
         k16, kD, SD, kDH,
         nullptr, nullptr, 0, 0, 0,
         nullptr, s16, kS, HSS, SS,
         kS, kS, kDH, kB * kH, kH, 0);
    // vT[b,h,dh,s] <- v (q16 region is dead now)
    vperm_kernel<<<(kM * kD) / 256, 256, 0, stream>>>(v16, vT);
    // softmax(scores / sqrt(DH)) in place
    softmax_kernel<<<kB * kH * kS, 256, 0, stream>>>(s16, 0.125f);
    // o[b,:,h,:] = P @ V[b,:,h,:]
    gemm(stream, s16, kS, HSS, SS,
         vT, kS, HDS, DS,
         nullptr, nullptr, 0, 0, 0,
         nullptr, o16, kD, SD, kDH,
         kS, kDH, kS, kB * kH, kH, 0);
    // h = LN(h + o @ Wo + bo)
    gemm(stream, o16, kD, 0, 0, woT + wOff, kD, 0, 0, bo + l * kD,
         hf, kD, 0, 0, tmpf, nullptr, kD, 0, 0, kM, kD, kD, 1, 1, 0);
    ln_kernel<<<kM, 256, 0, stream>>>(tmpf, ln1_s + l * kD, ln1_b + l * kD, hf, hh);
    // FFN
    gemm(stream, hh, kD, 0, 0, w1T + (size_t)l * kD * kDFF, kD, 0, 0, b1 + l * kDFF,
         nullptr, 0, 0, 0, nullptr, f1h, kDFF, 0, 0, kM, kDFF, kD, 1, 1, 1);
    gemm(stream, f1h, kDFF, 0, 0, w2T + (size_t)l * kDFF * kD, kDFF, 0, 0, b2 + l * kD,
         hf, kD, 0, 0, tmpf, nullptr, kD, 0, 0, kM, kD, kDFF, 1, 1, 0);
    ln_kernel<<<kM, 256, 0, stream>>>(tmpf, ln2_s + l * kD, ln2_b + l * kD, hf, hh);
  }

  // ---- bottleneck heads: z = relu(relu(h@e1)@e2 @ d1) @ d2 ----
  float* zf = out + 2 + kM;  // z written straight into d_out (b,s,d order)
  gemm(stream, hh, kD, 0, 0, e1T, kD, 0, 0, e1_b,
       nullptr, 0, 0, 0, nullptr, g1h, 128, 0, 0, kM, 128, kD, 1, 1, 1);
  gemm(stream, g1h, 128, 0, 0, e2T, 128, 0, 0, e2_b,
       nullptr, 0, 0, 0, nullptr, ench, kD, 0, 0, kM, kD, 128, 1, 1, 0);
  gemm(stream, ench, kD, 0, 0, d1T, kD, 0, 0, d1_b,
       nullptr, 0, 0, 0, nullptr, g2h, 128, 0, 0, kM, 128, kD, 1, 1, 1);
  gemm(stream, g2h, 128, 0, 0, d2T, 128, 0, 0, d2_b,
       nullptr, 0, 0, 0, zf, z16, kD, 0, 0, kM, kD, 128, 1, 1, 0);

  // ---- clustering epilogue ----
  init_kernel<<<1, 256, 0, stream>>>(out, cmbuf);
  // gram[s] = Z_s (64x512) @ Z_s^T via WMMA (batch over s; both operands K-contiguous)
  gemm(stream, z16, SD, kD, 0,
       z16, SD, kD, 0,
       nullptr, nullptr, 0, 0, 0,
       gram, nullptr, kB, (long)kB * kB, 0,
       kB, kB, kD, kS, 1, 0);
  knn_kernel<<<kM / 256, 256, 0, stream>>>(gram, y, assigned_i, out);
  cm_kernel<<<kM / 256, 256, 0, stream>>>(assigned_i, y, cmbuf);
  ari_kernel<<<1, 64, 0, stream>>>(cmbuf, out);
}
